// LiquidCell_35021163331814
// MI455X (gfx1250) — matrix operand order repaired and checked by reading, not compiled, not run on hardware
//
#include <hip/hip_runtime.h>

typedef __attribute__((ext_vector_type(16))) __bf16 v16bf;
typedef __attribute__((ext_vector_type(8)))  float  v8f;

#define B_    64
#define S_    1024
#define I_    256
#define H_    512
#define KTOT  768     // I_ + H_
#define NKT   24      // KTOT / 32
#define NNT   32      // H_ / 16
#define DT_   0.1f
#define LN_EPS_ 1e-5f

__device__ __forceinline__ unsigned short f2bf(float f) {
    unsigned int u = __float_as_uint(f);
    u += 0x7FFFu + ((u >> 16) & 1u);         // round-to-nearest-even
    return (unsigned short)(u >> 16);
}

// ---------------------------------------------------------------------------
// One-shot: pack W_in|W_rec (as Wf, K=768) and tau_w1 (as Wt, K=768) into bf16
// WMMA B-fragment order: [ntile][ktile][lane][e], value = W^T[k][n] = W[n][k]
// B 32x16 bf16 layout (ISA 7.12.2): lane L -> n = L&15, k = kt*32 + 16*(e>>3) + 8*(L>>4) + (e&7)
// ---------------------------------------------------------------------------
__global__ void pack_weights_kernel(const float* __restrict__ W_in,
                                    const float* __restrict__ W_rec,
                                    const float* __restrict__ tau_w1,
                                    unsigned short* __restrict__ wf,
                                    unsigned short* __restrict__ wt) {
    int idx = blockIdx.x * blockDim.x + threadIdx.x;
    if (idx >= NNT * NKT * 512) return;
    int within = idx & 511;
    int frag   = idx >> 9;
    int kt     = frag % NKT;
    int nt     = frag / NKT;
    int lane   = within >> 4;
    int e      = within & 15;
    int n = nt * 16 + (lane & 15);
    int k = kt * 32 + 16 * (e >> 3) + 8 * (lane >> 4) + (e & 7);
    float fv = (k < I_) ? W_in[n * I_ + k] : W_rec[n * H_ + (k - I_)];
    float tv = tau_w1[n * KTOT + k];
    wf[idx] = f2bf(fv);
    wt[idx] = f2bf(tv);
}

// ---------------------------------------------------------------------------
// Persistent recurrent kernel: 4 workgroups x 512 threads (16 wave32 each).
// Workgroup wg owns batch rows [wg*16, wg*16+16); loops all S=1024 steps.
// ---------------------------------------------------------------------------
__global__ __launch_bounds__(512, 1)
void liquid_cell_kernel(const float* __restrict__ x,
                        const float* __restrict__ h0,
                        const float* __restrict__ b_in,
                        const float* __restrict__ tau_b1,
                        const float* __restrict__ tau_w2,
                        const float* __restrict__ tau_b2,
                        const float* __restrict__ gamma,
                        const float* __restrict__ beta,
                        const unsigned short* __restrict__ wf,
                        const unsigned short* __restrict__ wt,
                        float* __restrict__ out,        // [B,S,H]
                        float* __restrict__ h_final,    // [B,H]
                        float* __restrict__ tau_hist) { // [B,S]
    // A-fragment buffer for [x_t | h] (16 rows x 768 K) in bf16, frag order.
    __shared__ unsigned short a_buf[NKT * 512];        // 24 KB
    __shared__ float h_plain[16 * H_];                 // 32 KB (f32 state)
    __shared__ float c_bin[H_], c_tb1[H_], c_tw2[H_], c_g[H_], c_b[H_];
    __shared__ float tau_part[16 * 16];
    __shared__ float tau_row[16];
    __shared__ float c_tb2;

    const int tid  = threadIdx.x;
    const int lane = tid & 31;
    const int wv   = tid >> 5;      // wave 0..15
    const int wg   = blockIdx.x;    // 0..3

    // ---- init: constants + h0 (f32 state + bf16 A-fragments) ----
    if (tid < H_) {
        c_bin[tid] = b_in[tid];
        c_tb1[tid] = tau_b1[tid];
        c_tw2[tid] = tau_w2[tid];
        c_g[tid]   = gamma[tid];
        c_b[tid]   = beta[tid];
    }
    if (tid == 0) c_tb2 = tau_b2[0];
    for (int idx = tid; idx < 16 * H_; idx += 512) {
        int m = idx >> 9;
        int n = idx & 511;
        float v = h0[(size_t)(wg * 16 + m) * H_ + n];
        h_plain[idx] = v;
        int kk = n & 31;
        int la = m + 16 * ((kk >> 3) & 1);
        int e  = ((kk >> 4) & 1) * 8 + (kk & 7);
        a_buf[(8 + (n >> 5)) * 512 + la * 16 + e] = f2bf(v);
    }
    __syncthreads();

    const int nt0 = 2 * wv;
    const int nt1 = 2 * wv + 1;
    const unsigned short* wfp0 = wf + (size_t)nt0 * NKT * 512 + lane * 16;
    const unsigned short* wfp1 = wf + (size_t)nt1 * NKT * 512 + lane * 16;
    const unsigned short* wtp0 = wt + (size_t)nt0 * NKT * 512 + lane * 16;
    const unsigned short* wtp1 = wt + (size_t)nt1 * NKT * 512 + lane * 16;

    for (int s = 0; s < S_; ++s) {
        // ---- phase A: x_t -> a_buf tiles 0..7 (bf16, A-fragment layout) ----
        {
            int m     = tid >> 5;                // 0..15 (row)
            int kbase = (tid & 31) * 8;          // 0..248, 8 consecutive k
            const float* xp = x + ((size_t)(wg * 16 + m) * S_ + s) * I_ + kbase;
            float4 v0 = *(const float4*)(xp);
            float4 v1 = *(const float4*)(xp + 4);
            if (s + 1 < S_) __builtin_prefetch(xp + I_, 0, 3);  // next step's x_t
            int kk = kbase & 31;
            int la = m + 16 * ((kk >> 3) & 1);
            int e0 = ((kk >> 4) & 1) * 8;        // kk&7 == 0
            unsigned short* dst = &a_buf[(kbase >> 5) * 512 + la * 16 + e0];
            dst[0] = f2bf(v0.x); dst[1] = f2bf(v0.y);
            dst[2] = f2bf(v0.z); dst[3] = f2bf(v0.w);
            dst[4] = f2bf(v1.x); dst[5] = f2bf(v1.y);
            dst[6] = f2bf(v1.z); dst[7] = f2bf(v1.w);
        }
        __syncthreads();

        // ---- phase B: both GEMMs via WMMA bf16, shared A fragment ----
        v8f fa0 = {}; v8f fa1 = {}; v8f ta0 = {}; v8f ta1 = {};
        #pragma unroll 4
        for (int kt = 0; kt < NKT; ++kt) {
            v16bf a   = *(const v16bf*)&a_buf[kt * 512 + lane * 16];
            v16bf bf0 = *(const v16bf*)(wfp0 + (size_t)kt * 512);
            v16bf bf1 = *(const v16bf*)(wfp1 + (size_t)kt * 512);
            v16bf bt0 = *(const v16bf*)(wtp0 + (size_t)kt * 512);
            v16bf bt1 = *(const v16bf*)(wtp1 + (size_t)kt * 512);
            fa0 = __builtin_amdgcn_wmma_f32_16x16x32_bf16(false, a, false, bf0, (short)0, fa0, false, false);
            fa1 = __builtin_amdgcn_wmma_f32_16x16x32_bf16(false, a, false, bf1, (short)0, fa1, false, false);
            ta0 = __builtin_amdgcn_wmma_f32_16x16x32_bf16(false, a, false, bt0, (short)0, ta0, false, false);
            ta1 = __builtin_amdgcn_wmma_f32_16x16x32_bf16(false, a, false, bt1, (short)0, ta1, false, false);
        }

        // ---- tau partials: tanh(t_hid)·tau_w2, reduce across N ----
        {
            int   c0  = nt0 * 16 + (lane & 15);
            int   c1  = nt1 * 16 + (lane & 15);
            float tb0 = c_tb1[c0], tw0 = c_tw2[c0];
            float tb1 = c_tb1[c1], tw1 = c_tw2[c1];
            float p[8];
            #pragma unroll
            for (int v = 0; v < 8; ++v)
                p[v] = tanhf(ta0[v] + tb0) * tw0 + tanhf(ta1[v] + tb1) * tw1;
            #pragma unroll
            for (int msk = 1; msk <= 8; msk <<= 1) {     // stays in 16-lane half
                #pragma unroll
                for (int v = 0; v < 8; ++v) p[v] += __shfl_xor(p[v], msk, 32);
            }
            if ((lane & 15) == 0) {
                int rbase = (lane >> 4) * 8;             // rows 0-7 / 8-15
                #pragma unroll
                for (int v = 0; v < 8; ++v) tau_part[wv * 16 + rbase + v] = p[v];
            }
        }
        __syncthreads();

        if (tid < 16) {
            float ssum = 0.f;
            #pragma unroll
            for (int w = 0; w < 16; ++w) ssum += tau_part[w * 16 + tid];
            float sig = 1.0f / (1.0f + __expf(-(ssum + c_tb2)));
            float tau = 1.0f + 9.0f * sig;               // TAU_MIN + range*sig
            tau_row[tid] = tau;
            tau_hist[(size_t)(wg * 16 + tid) * S_ + s] = tau;
        }
        __syncthreads();

        // ---- phase C: h_cand = h + DT*(tanh(f)+(-h))/tau, into h_plain ----
        {
            int   c0  = nt0 * 16 + (lane & 15);
            int   c1  = nt1 * 16 + (lane & 15);
            float bi0 = c_bin[c0], bi1 = c_bin[c1];
            int rbase = (lane >> 4) * 8;
            #pragma unroll
            for (int v = 0; v < 8; ++v) {
                int m = rbase + v;
                float dtau = DT_ / tau_row[m];
                float f0 = tanhf(fa0[v] + bi0);
                float f1 = tanhf(fa1[v] + bi1);
                float h0v = h_plain[m * H_ + c0];
                float h1v = h_plain[m * H_ + c1];
                h_plain[m * H_ + c0] = h0v + dtau * (f0 - h0v);
                h_plain[m * H_ + c1] = h1v + dtau * (f1 - h1v);
            }
        }
        __syncthreads();

        // ---- phase D: LayerNorm row m = wv; write out + refresh h frags ----
        {
            float vals[16];
            float s1 = 0.f, s2 = 0.f;
            #pragma unroll
            for (int j = 0; j < 16; ++j) {
                float v = h_plain[wv * H_ + lane + 32 * j];
                vals[j] = v; s1 += v; s2 += v * v;
            }
            #pragma unroll
            for (int msk = 1; msk <= 16; msk <<= 1) {
                s1 += __shfl_xor(s1, msk, 32);
                s2 += __shfl_xor(s2, msk, 32);
            }
            float mu  = s1 * (1.0f / H_);
            float var = s2 * (1.0f / H_) - mu * mu;
            float rs  = rsqrtf(var + LN_EPS_);
            float* orow = out + ((size_t)(wg * 16 + wv) * S_ + s) * H_;
            int la = wv + 16 * ((lane >> 3) & 1);
            int e  = ((lane >> 4) & 1) * 8 + (lane & 7);
            #pragma unroll
            for (int j = 0; j < 16; ++j) {
                int n = lane + 32 * j;
                float hv = (vals[j] - mu) * rs * c_g[n] + c_b[n];
                h_plain[wv * H_ + n] = hv;
                orow[n] = hv;
                a_buf[(8 + j) * 512 + la * 16 + e] = f2bf(hv);  // next-step A frag
            }
        }
        __syncthreads();
    }

    // ---- final hidden state ----
    for (int idx = tid; idx < 16 * H_; idx += 512) {
        int m = idx >> 9;
        int n = idx & 511;
        h_final[(size_t)(wg * 16 + m) * H_ + n] = h_plain[idx];
    }
}

// ---------------------------------------------------------------------------
extern "C" void kernel_launch(void* const* d_in, const int* in_sizes, int n_in,
                              void* d_out, int out_size, void* d_ws, size_t ws_size,
                              hipStream_t stream) {
    const float* x      = (const float*)d_in[0];
    const float* h0     = (const float*)d_in[1];
    const float* W_in   = (const float*)d_in[2];
    const float* b_in   = (const float*)d_in[3];
    const float* W_rec  = (const float*)d_in[4];
    const float* tau_w1 = (const float*)d_in[5];
    const float* tau_b1 = (const float*)d_in[6];
    const float* tau_w2 = (const float*)d_in[7];
    const float* tau_b2 = (const float*)d_in[8];
    const float* gamma  = (const float*)d_in[9];
    const float* beta   = (const float*)d_in[10];

    float* out      = (float*)d_out;
    float* h_final  = out + (size_t)B_ * S_ * H_;
    float* tau_hist = h_final + (size_t)B_ * H_;

    unsigned short* wf = (unsigned short*)d_ws;               // 768x512 bf16
    unsigned short* wt = wf + (size_t)NNT * NKT * 512;        // 768x512 bf16

    const int packN = NNT * NKT * 512;
    hipLaunchKernelGGL(pack_weights_kernel, dim3((packN + 255) / 256), dim3(256),
                       0, stream, W_in, W_rec, tau_w1, wf, wt);
    hipLaunchKernelGGL(liquid_cell_kernel, dim3(4), dim3(512), 0, stream,
                       x, h0, b_in, tau_b1, tau_w2, tau_b2, gamma, beta,
                       wf, wt, out, h_final, tau_hist);
}
